// SurvPLE_26809185861688
// MI455X (gfx1250) — matrix operand order repaired and checked by reading, not compiled, not run on hardware
//
#include <hip/hip_runtime.h>
#include <math.h>

typedef __attribute__((ext_vector_type(16))) _Float16 v16h;
typedef __attribute__((ext_vector_type(8)))  float    v8f;

#define TILE 4096
#define BLOCK 256
#define WAVES (BLOCK / 32)
#define ROWS_PER_BLOCK (WAVES * 16)

// risk_sum[i] = sum_j (|y[j]| >= |y[i]|) * exp(theta[j])
// Computed as tiled WMMA: A = 0/1 risk-set mask (f16, exact),
// B = exp(theta) split into hi/lo f16 halves, f32 accumulate.
__global__ __launch_bounds__(BLOCK)
void cox_risk_wmma(const float* __restrict__ y,
                   const float* __restrict__ yhat,
                   float* __restrict__ risk, int n) {
  __shared__ __align__(32) float    sT[TILE];
  __shared__ __align__(32) _Float16 sVhi[TILE];
  __shared__ __align__(32) _Float16 sVlo[TILE];

  const int lane = threadIdx.x & 31;
  const int wave = threadIdx.x >> 5;
  const int m    = lane & 15;   // A-matrix row within stripe
  const int h    = lane >> 4;   // half-wave selects K-group
  const int rowA = blockIdx.x * ROWS_PER_BLOCK + wave * 16 + m;

  const float Ti = (rowA < n) ? fabsf(y[rowA]) : __builtin_inff();

  v8f c = {0.f, 0.f, 0.f, 0.f, 0.f, 0.f, 0.f, 0.f};

  const int ntiles = (n + TILE - 1) / TILE;
  for (int t = 0; t < ntiles; ++t) {
    const int base = t * TILE;
    // Cooperative LDS fill: T = |y|, v = exp(theta) split hi/lo f16.
    for (int idx = (int)threadIdx.x; idx < TILE; idx += BLOCK) {
      const int j = base + idx;
      if (j < n) {
        sT[idx] = fabsf(y[j]);
        const float v = expf(yhat[j]);
        const _Float16 vh = (_Float16)v;
        sVhi[idx] = vh;
        sVlo[idx] = (_Float16)(v - (float)vh);
      } else {
        sT[idx]  = -__builtin_inff();      // mask = 0 for padded columns
        sVhi[idx] = (_Float16)0.f;
        sVlo[idx] = (_Float16)0.f;
      }
    }
    __syncthreads();

    // Each wave: one 16-row stripe, march K in steps of 32 columns.
    for (int c0 = 0; c0 < TILE; c0 += 32) {
      // A tile 16x32 f16: lane(m,h); element e -> K = 16h+e
      v16h a;
#pragma unroll
      for (int e = 0; e < 16; ++e) {
        const int K = 16 * h + e;
        a[e] = (sT[c0 + K] >= Ti) ? (_Float16)1.0f : (_Float16)0.0f;
      }
      // B tile 32x16 f16 (replicated across N): lane(n=lane%16, g=h);
      // element e -> K = 16*g + e  => 16 consecutive halves (32B LDS vector load)
      const v16h bh = *(const v16h*)(&sVhi[c0 + 16 * h]);
      const v16h bl = *(const v16h*)(&sVlo[c0 + 16 * h]);
      c = __builtin_amdgcn_wmma_f32_16x16x32_f16(false, a, false, bh,
                                                 (short)0, c, false, false);
      c = __builtin_amdgcn_wmma_f32_16x16x32_f16(false, a, false, bl,
                                                 (short)0, c, false, false);
    }
    __syncthreads();
  }

  // C/D layout: lane%16 = N (all N columns identical), VGPR r -> M = r + 8h.
  if (m == 0) {
#pragma unroll
    for (int r = 0; r < 8; ++r) {
      const int orow = blockIdx.x * ROWS_PER_BLOCK + wave * 16 + 8 * h + r;
      if (orow < n) risk[orow] = c[r];
    }
  }
}

// loss = -(1/n) * sum_i E_i * (theta_i - log(risk_sum_i)); deterministic
// single-block tree reduction (no float atomics).
__global__ __launch_bounds__(256)
void cox_loss_reduce(const float* __restrict__ y,
                     const float* __restrict__ yhat,
                     const float* __restrict__ risk,
                     float* __restrict__ out, int n) {
  __shared__ float s[256];
  float acc = 0.f;
  for (int i = (int)threadIdx.x; i < n; i += 256) {
    if (y[i] > 0.f) acc += yhat[i] - logf(risk[i]);
  }
  s[threadIdx.x] = acc;
  __syncthreads();
  for (int ofs = 128; ofs > 0; ofs >>= 1) {
    if ((int)threadIdx.x < ofs) s[threadIdx.x] += s[threadIdx.x + ofs];
    __syncthreads();
  }
  if (threadIdx.x == 0) out[0] = -s[0] / (float)n;
}

extern "C" void kernel_launch(void* const* d_in, const int* in_sizes, int n_in,
                              void* d_out, int out_size, void* d_ws, size_t ws_size,
                              hipStream_t stream) {
  (void)n_in; (void)out_size; (void)ws_size;
  const float* y    = (const float*)d_in[0];
  const float* yhat = (const float*)d_in[1];   // [n,1] flat
  float* risk = (float*)d_ws;                  // n floats of scratch
  float* out  = (float*)d_out;                 // single f32 loss
  const int n = in_sizes[0];

  const int blocks = (n + ROWS_PER_BLOCK - 1) / ROWS_PER_BLOCK;
  cox_risk_wmma<<<blocks, BLOCK, 0, stream>>>(y, yhat, risk, n);
  cox_loss_reduce<<<1, 256, 0, stream>>>(y, yhat, risk, out, n);
}
